// SelfAttentionLayer_27642409517774
// MI455X (gfx1250) — matrix operand order repaired and checked by reading, not compiled, not run on hardware
//
#include <hip/hip_runtime.h>
#include <hip/hip_bf16.h>
#include <stdint.h>

typedef __bf16 bf16;
typedef __attribute__((ext_vector_type(16))) __bf16 v16bf;
typedef __attribute__((ext_vector_type(8)))  float  v8f;
typedef __attribute__((ext_vector_type(4)))  unsigned uint4v;

#define BB_  2
#define S_   2048
#define D_   1024
#define NH_  16
#define HD_  64
#define DF_  4096

union ABu {
  v16bf  v;
  unsigned u[8];
  uint4v q[2];
};

#define WMMA_BF16(Am, Bm, Cm) \
  __builtin_amdgcn_wmma_f32_16x16x32_bf16(false, (Am), false, (Bm), (short)0, (Cm), false, false)

// 16-bit A-operand element-pair -> K mapping (pair j of the v16bf, h = lane/16)
__device__ __forceinline__ int amap(int j, int h) {
  return h * 16 + 2 * j;
}
__device__ __forceinline__ float rowmax16(float v) {
#pragma unroll
  for (int o = 1; o < 16; o <<= 1) v = fmaxf(v, __shfl_xor(v, o, 32));
  return v;
}
__device__ __forceinline__ float rowsum16(float v) {
#pragma unroll
  for (int o = 1; o < 16; o <<= 1) v += __shfl_xor(v, o, 32);
  return v;
}

// ---------------------------------------------------------------- elementwise
__global__ __launch_bounds__(256) void f32_to_bf16_kernel(const float* __restrict__ src,
                                                          bf16* __restrict__ dst, int n) {
  int i = blockIdx.x * 256 + threadIdx.x;
  if (i < n) dst[i] = (bf16)src[i];
}

// W[K][N] f32 -> WT[N][K] bf16, 32x32 LDS tile
__global__ __launch_bounds__(256) void transpose_cvt_kernel(const float* __restrict__ W,
                                                            bf16* __restrict__ WT, int K, int N) {
  __shared__ bf16 tile[32][33];
  const int nb = blockIdx.x * 32, kb = blockIdx.y * 32;
  const int tx = threadIdx.x & 31, ty = threadIdx.x >> 5;  // ty 0..7
#pragma unroll
  for (int i = 0; i < 32; i += 8)
    tile[ty + i][tx] = (bf16)W[(size_t)(kb + ty + i) * N + nb + tx];
  __syncthreads();
#pragma unroll
  for (int i = 0; i < 32; i += 8)
    WT[(size_t)(nb + ty + i) * K + kb + tx] = tile[tx][ty + i];
}

// V [B*S, D] bf16 -> VT [B*NH, HD, S] bf16 per-head transpose
__global__ __launch_bounds__(256) void transpose_v_kernel(const bf16* __restrict__ V,
                                                          bf16* __restrict__ VT) {
  __shared__ bf16 tile[32][33];
  const int bh = blockIdx.z;
  const int bb = bh / NH_, hh = bh % NH_;
  const int sb = blockIdx.x * 32, db = blockIdx.y * 32;
  const int tx = threadIdx.x & 31, ty = threadIdx.x >> 5;
#pragma unroll
  for (int i = 0; i < 32; i += 8)
    tile[ty + i][tx] = V[(size_t)(bb * S_ + sb + ty + i) * D_ + hh * HD_ + db + tx];
  __syncthreads();
#pragma unroll
  for (int i = 0; i < 32; i += 8)
    VT[((size_t)bh * HD_ + db + ty + i) * S_ + sb + tx] = tile[tx][ty + i];
}

// ---------------------------------------------------------------- WMMA GEMM
// C[M,N] = A[M,K](bf16) @ BT[N,K](bf16)^T  (+bias, +relu), out bf16 or f32.
// Block 256 thr = 8 waves arranged 2M x 4N. Wave tile 32x64 (2 A-ops, 8 accs).
// Block tile 64M x 256N. K-step 32.
template <bool HASBIAS, bool DORELU, bool OUTBF>
__global__ __launch_bounds__(256) void gemm_wmma_kernel(
    const bf16* __restrict__ A, const bf16* __restrict__ BT,
    const float* __restrict__ bias, float* __restrict__ Cf, bf16* __restrict__ Cb,
    int M, int N, int K) {
  __shared__ __align__(16) bf16 As[64][40];    // 64 M-rows x 32 K (+pad)
  __shared__ __align__(16) bf16 Bs[256][40];   // 256 N-rows x 32 K (+pad)
  const int tid = threadIdx.x;
  const int lane = tid & 31, w = tid >> 5;
  const int wm = w & 1, wn = w >> 1;           // 2 x 4 wave grid
  const int Mblk = blockIdx.y * 64, Nblk = blockIdx.x * 256;
  const int h = lane >> 4, ln = lane & 15;
  v8f acc[2][4] = {};
  const int srow = tid >> 2, scb = (tid & 3) * 8;  // 4 lanes/row, 16B each
  const bf16* agp = A + (size_t)(Mblk + srow) * K + scb;

  for (int k0 = 0; k0 < K; k0 += 32) {
    *(uint4v*)&As[srow][scb] = *(const uint4v*)(agp + k0);
#pragma unroll
    for (int rr = 0; rr < 4; ++rr) {  // B: 256 rows in 4 coalesced passes
      const int br = rr * 64 + srow;
      *(uint4v*)&Bs[br][scb] =
          *(const uint4v*)(BT + (size_t)(Nblk + br) * K + k0 + scb);
    }
    if (k0 + 32 < K) {  // warm next K tile (global_prefetch_b8)
      __builtin_prefetch(agp + k0 + 32, 0, 1);
      __builtin_prefetch(BT + (size_t)(Nblk + srow) * K + k0 + 32 + scb, 0, 1);
    }
    __syncthreads();
    ABu a0, a1;
#pragma unroll
    for (int j = 0; j < 8; ++j) {
      const int kk = amap(j, h);
      a0.u[j] = *(const unsigned*)&As[wm * 32 + ln][kk];
      a1.u[j] = *(const unsigned*)&As[wm * 32 + 16 + ln][kk];
    }
#pragma unroll
    for (int j = 0; j < 4; ++j) {
      const bf16* rb = &Bs[wn * 64 + j * 16 + ln][h * 16];
      ABu b;
      b.q[0] = *(const uint4v*)rb;
      b.q[1] = *(const uint4v*)(rb + 8);
      acc[0][j] = WMMA_BF16(a0.v, b.v, acc[0][j]);
      acc[1][j] = WMMA_BF16(a1.v, b.v, acc[1][j]);
    }
    __syncthreads();
  }
#pragma unroll
  for (int mi = 0; mi < 2; ++mi) {
#pragma unroll
    for (int j = 0; j < 4; ++j) {
      const int col = Nblk + wn * 64 + j * 16 + ln;
      const float bv = HASBIAS ? bias[col] : 0.f;
#pragma unroll
      for (int r = 0; r < 8; ++r) {
        const int rowi = Mblk + wm * 32 + mi * 16 + r + 8 * h;  // D-layout
        float vv = acc[mi][j][r] + bv;
        if (DORELU) vv = fmaxf(vv, 0.f);
        if (OUTBF) Cb[(size_t)rowi * N + col] = (bf16)vv;
        else       Cf[(size_t)rowi * N + col] = vv;
      }
    }
  }
}

// ---------------------------------------------------------------- attention
// One wave = 16 queries of one (b,h); streams keys 64/iter; flash softmax
// with one cross-lane reduction set per 64 keys.
__global__ __launch_bounds__(256) void attn_wmma_kernel(
    const bf16* __restrict__ Qb, const bf16* __restrict__ Kb,
    const bf16* __restrict__ VT, const float* __restrict__ Amask,
    float* __restrict__ Ctx) {
  __shared__ bf16 Pls[8][16][66];  // per-wave P repack (16 q x 64 keys, +pad)
  const int tid = threadIdx.x, lane = tid & 31, w = tid >> 5;
  const int h = lane >> 4, ln = lane & 15;
  const int bq = blockIdx.z, hh = blockIdx.y;
  const int qbase = blockIdx.x * 128 + w * 16;
  const float scale = 0.125f;  // 1/sqrt(64)

  ABu aQ0, aQ1;  // Q 16x64 as two 16x32 A-operands
  const bf16* qrow = Qb + (size_t)(bq * S_ + qbase + ln) * D_ + hh * HD_;
#pragma unroll
  for (int j = 0; j < 8; ++j) {
    const int kk = amap(j, h);
    aQ0.u[j] = *(const unsigned*)(qrow + kk);
    aQ1.u[j] = *(const unsigned*)(qrow + 32 + kk);
  }
  float m[8], lsum[8];
#pragma unroll
  for (int r = 0; r < 8; ++r) { m[r] = -1e30f; lsum[r] = 0.f; }
  v8f O[4] = {};

  for (int kb = 0; kb < S_; kb += 64) {
    v8f s[4] = {};
#pragma unroll
    for (int c = 0; c < 4; ++c) {  // 4 score tiles of 16 keys
      const bf16* kr =
          Kb + (size_t)(bq * S_ + kb + c * 16 + ln) * D_ + hh * HD_ + h * 16;
      ABu k0, k1;
      k0.q[0] = *(const uint4v*)kr;        k0.q[1] = *(const uint4v*)(kr + 8);
      k1.q[0] = *(const uint4v*)(kr + 32); k1.q[1] = *(const uint4v*)(kr + 40);
      s[c] = WMMA_BF16(aQ0.v, k0.v, s[c]);
      s[c] = WMMA_BF16(aQ1.v, k1.v, s[c]);
    }
    float mk[4];
#pragma unroll
    for (int c = 0; c < 4; ++c) mk[c] = Amask[bq * S_ + kb + c * 16 + ln];
#pragma unroll
    for (int r = 0; r < 8; ++r) {
      float sv[4];
#pragma unroll
      for (int c = 0; c < 4; ++c) sv[c] = s[c][r] * scale + mk[c];
      float bm = fmaxf(fmaxf(sv[0], sv[1]), fmaxf(sv[2], sv[3]));
      bm = rowmax16(bm);
      const float nm = fmaxf(m[r], bm);
      const float alpha = __expf(m[r] - nm);
      float ps = 0.f;
#pragma unroll
      for (int c = 0; c < 4; ++c) {
        const float p = __expf(sv[c] - nm);
        ps += p;
        Pls[w][r + 8 * h][c * 16 + ln] = (bf16)p;  // D-layout -> [row][key]
      }
      lsum[r] = lsum[r] * alpha + rowsum16(ps);
      m[r] = nm;
      O[0][r] *= alpha; O[1][r] *= alpha; O[2][r] *= alpha; O[3][r] *= alpha;
    }
    ABu aP0, aP1;  // re-read P in A-operand layout (same-wave DS dependency)
#pragma unroll
    for (int j = 0; j < 8; ++j) {
      const int kk = amap(j, h);
      aP0.u[j] = *(const unsigned*)&Pls[w][ln][kk];
      aP1.u[j] = *(const unsigned*)&Pls[w][ln][32 + kk];
    }
    const bf16* vb = VT + ((size_t)(bq * NH_ + hh) * HD_ + ln) * S_ + kb + h * 16;
#pragma unroll
    for (int j = 0; j < 4; ++j) {
      const bf16* p = vb + (size_t)j * 16 * S_;
      ABu v0, v1;
      v0.q[0] = *(const uint4v*)p;        v0.q[1] = *(const uint4v*)(p + 8);
      v1.q[0] = *(const uint4v*)(p + 32); v1.q[1] = *(const uint4v*)(p + 40);
      O[j] = WMMA_BF16(aP0.v, v0.v, O[j]);
      O[j] = WMMA_BF16(aP1.v, v1.v, O[j]);
    }
  }
#pragma unroll
  for (int r = 0; r < 8; ++r) {
    const float invl = 1.f / lsum[r];
    const size_t base = (size_t)(bq * S_ + qbase + r + 8 * h) * D_ + hh * HD_ + ln;
    Ctx[base + 0]  = O[0][r] * invl;
    Ctx[base + 16] = O[1][r] * invl;
    Ctx[base + 32] = O[2][r] * invl;
    Ctx[base + 48] = O[3][r] * invl;
  }
}

// ---------------------------------------------------------------- add + LN
__global__ __launch_bounds__(256) void add_ln_kernel(
    const float* __restrict__ X, const float* __restrict__ Yd,
    const float* __restrict__ g, const float* __restrict__ be,
    float* __restrict__ outf, bf16* __restrict__ outb) {
  __shared__ float red[256];
  const int row = blockIdx.x, t = threadIdx.x;
  const float* x = X + (size_t)row * D_;
  const float* y = Yd + (size_t)row * D_;
  float v[4];
  float s = 0.f;
#pragma unroll
  for (int i = 0; i < 4; ++i) { int c = t + i * 256; v[i] = x[c] + y[c]; s += v[i]; }
  red[t] = s; __syncthreads();
  for (int o = 128; o > 0; o >>= 1) { if (t < o) red[t] += red[t + o]; __syncthreads(); }
  const float mu = red[0] * (1.f / D_);
  __syncthreads();
  float s2 = 0.f;
#pragma unroll
  for (int i = 0; i < 4; ++i) { float d = v[i] - mu; s2 += d * d; }
  red[t] = s2; __syncthreads();
  for (int o = 128; o > 0; o >>= 1) { if (t < o) red[t] += red[t + o]; __syncthreads(); }
  const float inv = rsqrtf(red[0] * (1.f / D_) + 1e-5f);
#pragma unroll
  for (int i = 0; i < 4; ++i) {
    int c = t + i * 256;
    float o = (v[i] - mu) * inv * g[c] + be[c];
    outf[(size_t)row * D_ + c] = o;
    if (outb) outb[(size_t)row * D_ + c] = (bf16)o;
  }
}

// ---------------------------------------------------------------- driver
extern "C" void kernel_launch(void* const* d_in, const int* in_sizes, int n_in,
                              void* d_out, int out_size, void* d_ws, size_t ws_size,
                              hipStream_t stream) {
  const float* H     = (const float*)d_in[0];
  const float* Amask = (const float*)d_in[1];
  const float* Wq    = (const float*)d_in[2];
  const float* Wk    = (const float*)d_in[3];
  const float* Wv    = (const float*)d_in[4];
  const float* W1    = (const float*)d_in[5];
  const float* b1    = (const float*)d_in[6];
  const float* W2    = (const float*)d_in[7];
  const float* b2    = (const float*)d_in[8];
  const float* g1    = (const float*)d_in[9];
  const float* be1   = (const float*)d_in[10];
  const float* g2    = (const float*)d_in[11];
  const float* be2   = (const float*)d_in[12];
  float* out = (float*)d_out;
  char* ws = (char*)d_ws;
  const size_t MB = (size_t)1 << 20;

  bf16*  Hbf  = (bf16*)(ws + 0 * MB);    // 8 MB
  bf16*  WqT  = (bf16*)(ws + 8 * MB);    // 2 MB
  bf16*  WkT  = (bf16*)(ws + 10 * MB);   // 2 MB
  bf16*  WvT  = (bf16*)(ws + 12 * MB);   // 2 MB
  bf16*  W1T  = (bf16*)(ws + 14 * MB);   // 8 MB
  bf16*  W2T  = (bf16*)(ws + 22 * MB);   // 8 MB
  bf16*  Qb   = (bf16*)(ws + 30 * MB);   // 8 MB
  bf16*  Kb   = (bf16*)(ws + 38 * MB);   // 8 MB
  bf16*  Vb   = (bf16*)(ws + 46 * MB);   // 8 MB
  bf16*  VTt  = (bf16*)(ws + 54 * MB);   // 8 MB
  float* Ctx  = (float*)(ws + 62 * MB);  // 16 MB
  float* Fout = (float*)(ws + 78 * MB);  // 16 MB
  bf16*  Fbf  = (bf16*)(ws + 94 * MB);   // 8 MB
  bf16*  Yb   = (bf16*)(ws + 30 * MB);   // 32 MB (reuses dead Q/K/V/VT)
  float* Zf   = (float*)(ws + 62 * MB);  // 16 MB (reuses dead Ctx)

  const int Mrows = BB_ * S_;  // 4096

  f32_to_bf16_kernel<<<(Mrows * D_) / 256, 256, 0, stream>>>(H, Hbf, Mrows * D_);
  transpose_cvt_kernel<<<dim3(D_ / 32, D_ / 32), 256, 0, stream>>>(Wq, WqT, D_, D_);
  transpose_cvt_kernel<<<dim3(D_ / 32, D_ / 32), 256, 0, stream>>>(Wk, WkT, D_, D_);
  transpose_cvt_kernel<<<dim3(D_ / 32, D_ / 32), 256, 0, stream>>>(Wv, WvT, D_, D_);
  transpose_cvt_kernel<<<dim3(DF_ / 32, D_ / 32), 256, 0, stream>>>(W1, W1T, D_, DF_);
  transpose_cvt_kernel<<<dim3(D_ / 32, DF_ / 32), 256, 0, stream>>>(W2, W2T, DF_, D_);

  gemm_wmma_kernel<false, false, true><<<dim3(D_ / 256, Mrows / 64), 256, 0, stream>>>(
      Hbf, WqT, nullptr, nullptr, Qb, Mrows, D_, D_);
  gemm_wmma_kernel<false, false, true><<<dim3(D_ / 256, Mrows / 64), 256, 0, stream>>>(
      Hbf, WkT, nullptr, nullptr, Kb, Mrows, D_, D_);
  gemm_wmma_kernel<false, false, true><<<dim3(D_ / 256, Mrows / 64), 256, 0, stream>>>(
      Hbf, WvT, nullptr, nullptr, Vb, Mrows, D_, D_);

  transpose_v_kernel<<<dim3(S_ / 32, HD_ / 32, BB_ * NH_), 256, 0, stream>>>(Vb, VTt);

  attn_wmma_kernel<<<dim3(S_ / 128, NH_, BB_), 256, 0, stream>>>(Qb, Kb, VTt, Amask, Ctx);

  add_ln_kernel<<<Mrows, 256, 0, stream>>>(H, Ctx, g1, be1, Fout, Fbf);

  gemm_wmma_kernel<true, true, true><<<dim3(DF_ / 256, Mrows / 64), 256, 0, stream>>>(
      Fbf, W1T, b1, nullptr, Yb, Mrows, DF_, D_);
  gemm_wmma_kernel<true, false, false><<<dim3(D_ / 256, Mrows / 64), 256, 0, stream>>>(
      Yb, W2T, b2, Zf, nullptr, Mrows, D_, DF_);

  add_ln_kernel<<<Mrows, 256, 0, stream>>>(Fout, Zf, g2, be2, out, nullptr);
}
